// WRATBlock_24567212933783
// MI455X (gfx1250) — compile-verified, matrix operand order checked by reading, not run on hardware
//
#include <hip/hip_runtime.h>
#include <cstdint>
#include <cstddef>

// ---------------------------------------------------------------------------
// Types
// ---------------------------------------------------------------------------
typedef __bf16 bf16_t;
typedef __attribute__((ext_vector_type(16))) __bf16 v16bf;
typedef __attribute__((ext_vector_type(8)))  float  v8f;
typedef __attribute__((ext_vector_type(4)))  unsigned int v4u_t;
typedef __attribute__((ext_vector_type(8)))  int    v8i_t;
typedef __attribute__((ext_vector_type(4)))  int    v4i_t;

#define B_  8
#define D_  512
#define L_  1024
#define H_  8
#define DH_ 64
#define N_  (B_ * L_)
#define D4_ (4 * D_)

__device__ __forceinline__ v8f v8f_zero() {
  v8f z = {0.f, 0.f, 0.f, 0.f, 0.f, 0.f, 0.f, 0.f};
  return z;
}

__device__ __forceinline__ float gelu_exact(float x) {
  return 0.5f * x * (1.0f + erff(x * 0.70710678118654752440f));
}

// Low 32 bits of a generic pointer to __shared__ = LDS byte address
// (addrspace(3)->generic cast is {shared_base_hi, lds_offset}).
__device__ __forceinline__ uint32_t lds_addr32(const void* p) {
  return (uint32_t)(uintptr_t)p;
}

// ---------------------------------------------------------------------------
// [B,D,L] f32  ->  [B*L, D] f32 + bf16   (token-major)
// ---------------------------------------------------------------------------
__global__ __launch_bounds__(256) void k_transpose_in(
    const float* __restrict__ src, float* __restrict__ dstF,
    bf16_t* __restrict__ dstB) {
  int idx = blockIdx.x * 256 + threadIdx.x;      // over N_*D_, output-major
  if (idx >= N_ * D_) return;
  int d = idx % D_;
  int t = idx / D_;                              // b*L + l
  int l = t % L_;
  int b = t / L_;
  float v = src[((size_t)b * D_ + d) * L_ + l];
  dstF[idx] = v;
  dstB[idx] = (bf16_t)v;
}

// [B*L, D] f32 -> [B,D,L] f32
__global__ __launch_bounds__(256) void k_transpose_out(
    const float* __restrict__ src, float* __restrict__ dst) {
  int idx = blockIdx.x * 256 + threadIdx.x;      // over B_*D_*L_, output-major
  if (idx >= B_ * D_ * L_) return;
  int l = idx % L_;
  int r = idx / L_;
  int d = r % D_;
  int b = r / D_;
  dst[idx] = src[((size_t)b * L_ + l) * D_ + d];
}

__global__ __launch_bounds__(256) void k_f32_to_bf16(
    const float* __restrict__ s, bf16_t* __restrict__ d, int n) {
  int i = blockIdx.x * 256 + threadIdx.x;
  if (i < n) d[i] = (bf16_t)s[i];
}

// keep[b*L+l] = (mean_d |lh_s[t,d]| > sigmoid(raw_tau)) ? 1 : 0
__global__ __launch_bounds__(256) void k_keep(
    const float* __restrict__ xlh, const float* __restrict__ raw_tau,
    float* __restrict__ keep) {
  int t = blockIdx.x * 256 + threadIdx.x;
  if (t >= N_) return;
  const float* row = xlh + (size_t)t * D_;
  float s = 0.f;
  for (int d = 0; d < D_; ++d) s += fabsf(row[d]);
  float tau = 1.0f / (1.0f + __expf(-raw_tau[0]));
  keep[t] = (s * (1.0f / (float)D_) > tau) ? 1.0f : 0.0f;
}

// ---------------------------------------------------------------------------
// WMMA bf16 GEMM: C[M,Nc] = act(A[M,K] @ W[K,Nc] + bias) + resid
// Block tile 128x128, 256 threads = 8 waves (2x4), wave tile 64x32 (4x2 WMMA).
// A tile (128x32, contiguous 2D tile of row-major A) is DMA'd by the Tensor
// Data Mover: wave 0 issues tensor_load_to_lds, overlapped with the W-tile
// transpose done by all threads, drained with s_wait_tensorcnt.
// ---------------------------------------------------------------------------
__global__ __launch_bounds__(256) void k_gemm_bf16(
    const bf16_t* __restrict__ A, const bf16_t* __restrict__ W,
    const float* __restrict__ bias, const float* __restrict__ resid,
    float* __restrict__ Cf, bf16_t* __restrict__ Cb,
    int M, int Nc, int K, int act) {
  __shared__ __align__(16) bf16_t sA[128][32];   // [m][k]  (TDM destination)
  __shared__ __align__(16) bf16_t sBt[128][48];  // [n][k], padded stride

  int tid  = threadIdx.x;
  int lane = tid & 31;
  int wave = tid >> 5;
  int wm = (wave >> 2) * 64;
  int wn = (wave & 3) * 32;
  int bm = blockIdx.y * 128;
  int bn = blockIdx.x * 128;
  int g   = lane >> 4;
  int l16 = lane & 15;

  v8f acc[4][2];
#pragma unroll
  for (int mt = 0; mt < 4; ++mt)
#pragma unroll
    for (int nt = 0; nt < 2; ++nt) acc[mt][nt] = v8f_zero();

  for (int k0 = 0; k0 < K; k0 += 32) {
    // --- A tile via TDM (one DMA for the whole 128x32 bf16 tile) ---
    if (wave == 0) {
      uint64_t ga  = (uint64_t)(uintptr_t)(A + (size_t)bm * K + k0);
      uint32_t lds = lds_addr32(&sA[0][0]);
      v4u_t g0;
      g0.x = 1u;                                    // count=1 (valid D#)
      g0.y = lds;                                   // lds_addr
      g0.z = (uint32_t)ga;                          // global_addr[31:0]
      g0.w = (uint32_t)((ga >> 32) & 0x01FFFFFFu)   // global_addr[56:32]
             | 0x80000000u;                         // type=2 ("image")
      v8i_t g1;
      g1[0] = (int)(1u << 16);                      // data_size=1 -> 2 bytes
      g1[1] = (int)(((uint32_t)K & 0xFFFFu) << 16); // tensor_dim0[15:0]
      g1[2] = (int)((((uint32_t)K >> 16) & 0xFFFFu) |
                    (((uint32_t)M & 0xFFFFu) << 16));   // dim0 hi | dim1 lo
      g1[3] = (int)((((uint32_t)M >> 16) & 0xFFFFu) |
                    (32u << 16));                   // dim1 hi | tile_dim0=32
      g1[4] = (int)128u;                            // tile_dim1=128, tile_dim2=0
      g1[5] = (int)(uint32_t)K;                     // tensor_dim0_stride lo
      g1[6] = 0;                                    // stride hi | dim1_stride lo
      g1[7] = 0;
      v4i_t gz4 = {0, 0, 0, 0};
      v8i_t gz8 = {0, 0, 0, 0, 0, 0, 0, 0};
      // 6-arg form on this toolchain: (g0, g1, g2, g3, g4, cpol)
      __builtin_amdgcn_tensor_load_to_lds(g0, g1, gz4, gz4, gz8, 0);
    }
    // --- stage W tile transposed: [k 32][n 128] -> sBt[n][k] ---
    {
      int r  = tid >> 3;            // k row 0..31
      int c0 = (tid & 7) * 16;      // n col base
      const bf16_t* src = W + (size_t)(k0 + r) * Nc + bn + c0;
#pragma unroll
      for (int i = 0; i < 16; ++i) sBt[c0 + i][r] = src[i];
      if (k0 + 32 < K) __builtin_prefetch(src + (size_t)32 * Nc, 0, 1);
    }
    if (wave == 0) __builtin_amdgcn_s_wait_tensorcnt(0);
    __syncthreads();

    v16bf bfr[2];
#pragma unroll
    for (int nt = 0; nt < 2; ++nt) {
      int n = wn + nt * 16 + l16;
      ((uint4*)&bfr[nt])[0] = *(const uint4*)&sBt[n][g * 8];
      ((uint4*)&bfr[nt])[1] = *(const uint4*)&sBt[n][16 + g * 8];
    }
#pragma unroll
    for (int mt = 0; mt < 4; ++mt) {
      v16bf af;
      int r = wm + mt * 16 + l16;
      ((uint4*)&af)[0] = *(const uint4*)&sA[r][g * 8];
      ((uint4*)&af)[1] = *(const uint4*)&sA[r][16 + g * 8];
#pragma unroll
      for (int nt = 0; nt < 2; ++nt)
        acc[mt][nt] = __builtin_amdgcn_wmma_f32_16x16x32_bf16(
            false, af, false, bfr[nt], (short)0, acc[mt][nt], false, false);
    }
    __syncthreads();
  }

  // epilogue (C layout: VGPR j -> row mo+j, lane%16 -> col)
  int mo = g * 8;
#pragma unroll
  for (int mt = 0; mt < 4; ++mt)
#pragma unroll
    for (int nt = 0; nt < 2; ++nt)
#pragma unroll
      for (int j = 0; j < 8; ++j) {
        int row = bm + wm + mt * 16 + mo + j;
        int col = bn + wn + nt * 16 + l16;
        float v = acc[mt][nt][j];
        if (bias)  v += bias[col];
        if (act)   v = gelu_exact(v);
        size_t o = (size_t)row * Nc + col;
        if (resid) v += resid[o];
        if (Cf) Cf[o] = v;
        if (Cb) Cb[o] = (bf16_t)v;
      }
}

// ---------------------------------------------------------------------------
// Flash attention, one (b,h) x 64-query tile per block, 4 waves.
// Wave w owns queries [q0+16w, q0+16w+16). Dh = 64.
// K tile is staged with async global->LDS copies (ASYNCcnt), V is staged
// transposed manually, P goes through per-wave LDS into A-fragment layout.
// ---------------------------------------------------------------------------
__global__ __launch_bounds__(128) void k_flash_attn(
    const bf16_t* __restrict__ Q, const bf16_t* __restrict__ Kp,
    const bf16_t* __restrict__ V, const float* __restrict__ keep,
    bf16_t* __restrict__ O) {
  __shared__ __align__(16) bf16_t sK[64][64];        // [key][dh]  (== B^T for S)
  __shared__ __align__(16) bf16_t sVt[64][64];       // [dh][key]  (== B^T for PV)
  __shared__ __align__(16) bf16_t sP[4][16][64];     // per-wave P tile

  int tid  = threadIdx.x;
  int lane = tid & 31;
  int wave = tid >> 5;
  int g   = lane >> 4;
  int l16 = lane & 15;
  int mo  = g * 8;

  int bh = blockIdx.y;
  int b  = bh / H_;
  int h  = bh % H_;
  int q0 = blockIdx.x * 64;

  // Q fragments straight from global (row-major token x Dh slice)
  const bf16_t* qrow = Q + ((size_t)b * L_ + q0 + wave * 16 + l16) * D_ + h * DH_;
  v16bf qf[2];
#pragma unroll
  for (int kk = 0; kk < 2; ++kk) {
    ((uint4*)&qf[kk])[0] = *(const uint4*)(qrow + kk * 32 + g * 8);
    ((uint4*)&qf[kk])[1] = *(const uint4*)(qrow + kk * 32 + 16 + g * 8);
  }

  v8f oacc[4];
#pragma unroll
  for (int nt = 0; nt < 4; ++nt) oacc[nt] = v8f_zero();
  float mrun[8], lrun[8];
#pragma unroll
  for (int j = 0; j < 8; ++j) { mrun[j] = -1e30f; lrun[j] = 0.f; }

  const float scale = 0.125f;  // 1/sqrt(64)

  for (int kb = 0; kb < L_; kb += 64) {
    int key = tid >> 1, c = (tid & 1) * 32;
    // --- K tile: async global->LDS, 64B per thread as 4 x b128.
    //     IOFFSET applies to both LDS dest and global source (ISA 10.x). ---
    {
      const bf16_t* ks = Kp + ((size_t)b * L_ + kb + key) * D_ + h * DH_ + c;
      uint32_t ldsk = lds_addr32(&sK[key][c]);
      asm volatile(
          "global_load_async_to_lds_b128 %0, %1, off\n\t"
          "global_load_async_to_lds_b128 %0, %1, off offset:16\n\t"
          "global_load_async_to_lds_b128 %0, %1, off offset:32\n\t"
          "global_load_async_to_lds_b128 %0, %1, off offset:48"
          :: "v"(ldsk), "v"(ks) : "memory");
    }
    // --- V tile transposed: [dh][key] (needs element transpose, manual) ---
    {
      const bf16_t* vs = V + ((size_t)b * L_ + kb + key) * D_ + h * DH_ + c;
#pragma unroll
      for (int i = 0; i < 32; ++i) sVt[c + i][key] = vs[i];
    }
    asm volatile("s_wait_asynccnt 0" ::: "memory");
    __syncthreads();

    // S = Q @ K^T  (K dim = Dh = 64, two WMMA k-steps)
    v8f sacc[4];
#pragma unroll
    for (int nt = 0; nt < 4; ++nt) sacc[nt] = v8f_zero();
#pragma unroll
    for (int kk = 0; kk < 2; ++kk) {
#pragma unroll
      for (int nt = 0; nt < 4; ++nt) {
        v16bf kf;
        int n = nt * 16 + l16;
        ((uint4*)&kf)[0] = *(const uint4*)&sK[n][kk * 32 + g * 8];
        ((uint4*)&kf)[1] = *(const uint4*)&sK[n][kk * 32 + 16 + g * 8];
        sacc[nt] = __builtin_amdgcn_wmma_f32_16x16x32_bf16(
            false, qf[kk], false, kf, (short)0, sacc[nt], false, false);
      }
    }

    // scale + energy mask
    float sv[4][8];
#pragma unroll
    for (int nt = 0; nt < 4; ++nt) {
      float kv = keep ? keep[(size_t)b * L_ + kb + nt * 16 + l16] : 1.0f;
#pragma unroll
      for (int j = 0; j < 8; ++j) {
        float x = sacc[nt][j] * scale;
        sv[nt][j] = (kv != 0.0f) ? x : -1e30f;
      }
    }

    // online softmax per row (rows live per 16-lane half; butterfly xor 1..8)
#pragma unroll
    for (int j = 0; j < 8; ++j) {
      float rm = fmaxf(fmaxf(sv[0][j], sv[1][j]), fmaxf(sv[2][j], sv[3][j]));
#pragma unroll
      for (int off = 1; off < 16; off <<= 1)
        rm = fmaxf(rm, __shfl_xor(rm, off, 32));
      float mnew = fmaxf(mrun[j], rm);
      float sc = (mnew <= -1e29f) ? 1.0f : __expf(mrun[j] - mnew);
      float rs = 0.f;
#pragma unroll
      for (int nt = 0; nt < 4; ++nt) {
        float p = (sv[nt][j] <= -1e29f) ? 0.0f : __expf(sv[nt][j] - mnew);
        rs += p;
        sP[wave][mo + j][nt * 16 + l16] = (bf16_t)p;
      }
#pragma unroll
      for (int off = 1; off < 16; off <<= 1) rs += __shfl_xor(rs, off, 32);
      lrun[j] = lrun[j] * sc + rs;
      mrun[j] = mnew;
#pragma unroll
      for (int nt = 0; nt < 4; ++nt) oacc[nt][j] = oacc[nt][j] * sc;
    }

    // O += P @ V  (intra-wave LDS: DS ops are in-order within a wave)
#pragma unroll
    for (int kk = 0; kk < 2; ++kk) {
      v16bf pf;
      ((uint4*)&pf)[0] = *(const uint4*)&sP[wave][l16][kk * 32 + g * 8];
      ((uint4*)&pf)[1] = *(const uint4*)&sP[wave][l16][kk * 32 + 16 + g * 8];
#pragma unroll
      for (int nt = 0; nt < 4; ++nt) {
        v16bf vf;
        int n = nt * 16 + l16;
        ((uint4*)&vf)[0] = *(const uint4*)&sVt[n][kk * 32 + g * 8];
        ((uint4*)&vf)[1] = *(const uint4*)&sVt[n][kk * 32 + 16 + g * 8];
        oacc[nt] = __builtin_amdgcn_wmma_f32_16x16x32_bf16(
            false, pf, false, vf, (short)0, oacc[nt], false, false);
      }
    }
    __syncthreads();
  }

  // normalize + write (all-masked rows -> 0, matching nan_to_num)
#pragma unroll
  for (int j = 0; j < 8; ++j) {
    float inv = (lrun[j] > 0.0f) ? 1.0f / lrun[j] : 0.0f;
    int tok = q0 + wave * 16 + mo + j;
#pragma unroll
    for (int nt = 0; nt < 4; ++nt)
      O[((size_t)b * L_ + tok) * D_ + h * DH_ + nt * 16 + l16] =
          (bf16_t)(oacc[nt][j] * inv);
  }
}

// ---------------------------------------------------------------------------
// LayerNorm over D, wave-per-token, up to 3 summed inputs.
// ---------------------------------------------------------------------------
__global__ __launch_bounds__(256) void k_ln(
    const float* __restrict__ x0, const float* __restrict__ x1,
    const float* __restrict__ x2, const float* __restrict__ gam,
    const float* __restrict__ bet, float* __restrict__ outF,
    bf16_t* __restrict__ outB) {
  int lane = threadIdx.x & 31;
  int wave = threadIdx.x >> 5;
  int t = blockIdx.x * 8 + wave;
  size_t base = (size_t)t * D_;

  float v[16];
#pragma unroll
  for (int i = 0; i < 16; ++i) {
    int d = lane + i * 32;
    float x = x0[base + d];
    if (x1) x += x1[base + d];
    if (x2) x += x2[base + d];
    v[i] = x;
  }
  float s = 0.f;
#pragma unroll
  for (int i = 0; i < 16; ++i) s += v[i];
#pragma unroll
  for (int off = 16; off >= 1; off >>= 1) s += __shfl_xor(s, off, 32);
  float mean = s * (1.0f / (float)D_);
  float q = 0.f;
#pragma unroll
  for (int i = 0; i < 16; ++i) { float dd = v[i] - mean; q += dd * dd; }
#pragma unroll
  for (int off = 16; off >= 1; off >>= 1) q += __shfl_xor(q, off, 32);
  float rstd = rsqrtf(q * (1.0f / (float)D_) + 1e-5f);
#pragma unroll
  for (int i = 0; i < 16; ++i) {
    int d = lane + i * 32;
    float y = (v[i] - mean) * rstd * gam[d] + bet[d];
    outF[base + d] = y;
    outB[base + d] = (bf16_t)y;
  }
}

// ---------------------------------------------------------------------------
// Host orchestration
// ---------------------------------------------------------------------------
extern "C" void kernel_launch(void* const* d_in, const int* in_sizes, int n_in,
                              void* d_out, int out_size, void* d_ws,
                              size_t ws_size, hipStream_t stream) {
  (void)in_sizes; (void)n_in; (void)out_size; (void)ws_size;

  enum {
    iLL = 0, iLH = 1, iTAU = 2,
    iLLQW = 3, iLLQB, iLLKW, iLLKB, iLLVW, iLLVB, iLLOW, iLLOB,
    iLHQW, iLHQB, iLHKW, iLHKB, iLHVW, iLHVB, iLHOW, iLHOB,
    iCRQW, iCRQB, iCRKW, iCRKB, iCRVW, iCRVB, iCROW, iCROB,
    iMLLW1, iMLLB1, iMLLW2, iMLLB2, iMLHW1, iMLHB1, iMLHW2, iMLHB2,
    iN1G, iN1B, iN2G, iN2B
  };

  const float* LL  = (const float*)d_in[iLL];
  const float* LHp = (const float*)d_in[iLH];
  const float* tau = (const float*)d_in[iTAU];

  char* ws = (char*)d_ws;
  size_t off = 0;
  auto alloc = [&](size_t bytes) -> void* {
    void* p = ws + off;
    off = (off + bytes + 255) & ~(size_t)255;
    return p;
  };

  const size_t ND = (size_t)N_ * D_;
  float*  Xll_f  = (float*)alloc(ND * 4);
  float*  Xlh_f  = (float*)alloc(ND * 4);
  bf16_t* Xll_b  = (bf16_t*)alloc(ND * 2);
  bf16_t* Xlh_b  = (bf16_t*)alloc(ND * 2);
  bf16_t* Qb     = (bf16_t*)alloc(ND * 2);
  bf16_t* Kb     = (bf16_t*)alloc(ND * 2);
  bf16_t* Vb     = (bf16_t*)alloc(ND * 2);
  bf16_t* CTXb   = (bf16_t*)alloc(ND * 2);
  float*  ll_o_f = (float*)alloc(ND * 4);
  bf16_t* ll_o_b = (bf16_t*)alloc(ND * 2);
  float*  lh_o_f = (float*)alloc(ND * 4);
  bf16_t* lh_o_b = (bf16_t*)alloc(ND * 2);
  float*  cr_o_f = (float*)alloc(ND * 4);
  float*  keep   = (float*)alloc((size_t)N_ * 4);
  float*  ln_f   = (float*)alloc(ND * 4);
  bf16_t* ln_b   = (bf16_t*)alloc(ND * 2);
  bf16_t* hid_b  = (bf16_t*)alloc((size_t)N_ * D4_ * 2);
  float*  fin_f  = (float*)alloc(ND * 4);

  // weight bf16 conversions (12 DxD, 4 MLP matrices)
  const int    widx[16] = { iLLQW, iLLKW, iLLVW, iLLOW, iLHQW, iLHKW, iLHVW,
                            iLHOW, iCRQW, iCRKW, iCRVW, iCROW, iMLLW1, iMLLW2,
                            iMLHW1, iMLHW2 };
  const size_t wcnt[16] = { (size_t)D_ * D_, (size_t)D_ * D_, (size_t)D_ * D_,
                            (size_t)D_ * D_, (size_t)D_ * D_, (size_t)D_ * D_,
                            (size_t)D_ * D_, (size_t)D_ * D_, (size_t)D_ * D_,
                            (size_t)D_ * D_, (size_t)D_ * D_, (size_t)D_ * D_,
                            (size_t)D_ * D4_, (size_t)D4_ * D_,
                            (size_t)D_ * D4_, (size_t)D4_ * D_ };
  bf16_t* Wb[16];
  for (int i = 0; i < 16; ++i) {
    Wb[i] = (bf16_t*)alloc(wcnt[i] * 2);
    int n = (int)wcnt[i];
    k_f32_to_bf16<<<(n + 255) / 256, 256, 0, stream>>>(
        (const float*)d_in[widx[i]], Wb[i], n);
  }

  int tblocks = (N_ * D_ + 255) / 256;
  k_transpose_in<<<tblocks, 256, 0, stream>>>(LL, Xll_f, Xll_b);
  k_transpose_in<<<tblocks, 256, 0, stream>>>(LHp, Xlh_f, Xlh_b);
  k_keep<<<(N_ + 255) / 256, 256, 0, stream>>>(Xlh_f, tau, keep);

  auto gemm = [&](const bf16_t* A, const bf16_t* W, const float* bias,
                  const float* resid, float* Cf, bf16_t* Cb, int M, int Nc,
                  int K, int act) {
    dim3 grid(Nc / 128, M / 128);
    k_gemm_bf16<<<grid, 256, 0, stream>>>(A, W, bias, resid, Cf, Cb, M, Nc, K,
                                          act);
  };
  dim3 fgrid(L_ / 64, B_ * H_);

  // --- ll self-attention ---
  gemm(Xll_b, Wb[0], (const float*)d_in[iLLQB], nullptr, nullptr, Qb, N_, D_, D_, 0);
  gemm(Xll_b, Wb[1], (const float*)d_in[iLLKB], nullptr, nullptr, Kb, N_, D_, D_, 0);
  gemm(Xll_b, Wb[2], (const float*)d_in[iLLVB], nullptr, nullptr, Vb, N_, D_, D_, 0);
  k_flash_attn<<<fgrid, 128, 0, stream>>>(Qb, Kb, Vb, nullptr, CTXb);
  gemm(CTXb, Wb[3], (const float*)d_in[iLLOB], nullptr, ll_o_f, ll_o_b, N_, D_, D_, 0);

  // --- lh self-attention (energy-masked) ---
  gemm(Xlh_b, Wb[4], (const float*)d_in[iLHQB], nullptr, nullptr, Qb, N_, D_, D_, 0);
  gemm(Xlh_b, Wb[5], (const float*)d_in[iLHKB], nullptr, nullptr, Kb, N_, D_, D_, 0);
  gemm(Xlh_b, Wb[6], (const float*)d_in[iLHVB], nullptr, nullptr, Vb, N_, D_, D_, 0);
  k_flash_attn<<<fgrid, 128, 0, stream>>>(Qb, Kb, Vb, keep, CTXb);
  gemm(CTXb, Wb[7], (const float*)d_in[iLHOB], nullptr, lh_o_f, lh_o_b, N_, D_, D_, 0);

  // --- cross attention: Q from ll_o, K/V from lh_o ---
  gemm(ll_o_b, Wb[8],  (const float*)d_in[iCRQB], nullptr, nullptr, Qb, N_, D_, D_, 0);
  gemm(lh_o_b, Wb[9],  (const float*)d_in[iCRKB], nullptr, nullptr, Kb, N_, D_, D_, 0);
  gemm(lh_o_b, Wb[10], (const float*)d_in[iCRVB], nullptr, nullptr, Vb, N_, D_, D_, 0);
  k_flash_attn<<<fgrid, 128, 0, stream>>>(Qb, Kb, Vb, nullptr, CTXb);
  gemm(CTXb, Wb[11], (const float*)d_in[iCROB], nullptr, cr_o_f, nullptr, N_, D_, D_, 0);

  // --- ll branch: LN(ll_s + ll_o + cr_o), MLP + residual, transpose out ---
  k_ln<<<N_ / 8, 256, 0, stream>>>(Xll_f, ll_o_f, cr_o_f,
                                   (const float*)d_in[iN1G],
                                   (const float*)d_in[iN1B], ln_f, ln_b);
  gemm(ln_b, Wb[12], (const float*)d_in[iMLLB1], nullptr, nullptr, hid_b, N_, D4_, D_, 1);
  gemm(hid_b, Wb[13], (const float*)d_in[iMLLB2], ln_f, fin_f, nullptr, N_, D_, D4_, 0);
  k_transpose_out<<<tblocks, 256, 0, stream>>>(fin_f, (float*)d_out);

  // --- lh branch: LN(lh_s + lh_o), MLP + residual, transpose out ---
  k_ln<<<N_ / 8, 256, 0, stream>>>(Xlh_f, lh_o_f, nullptr,
                                   (const float*)d_in[iN2G],
                                   (const float*)d_in[iN2B], ln_f, ln_b);
  gemm(ln_b, Wb[14], (const float*)d_in[iMLHB1], nullptr, nullptr, hid_b, N_, D4_, D_, 1);
  gemm(hid_b, Wb[15], (const float*)d_in[iMLHB2], ln_f, fin_f, nullptr, N_, D_, D4_, 0);
  k_transpose_out<<<tblocks, 256, 0, stream>>>(
      fin_f, (float*)d_out + (size_t)B_ * D_ * L_);
}
